// Encoder_55293408968979
// MI455X (gfx1250) — matrix operand order repaired and checked by reading, not compiled, not run on hardware
//
#include <hip/hip_runtime.h>

// ---------------------------------------------------------------------------
// Transformer encoder (6 layers) for MI455X / gfx1250, wave32 + WMMA.
// GEMMs run in f16 via V_WMMA_F32_16X16X32_F16 with f32 accumulation.
// Residual / LayerNorm / softmax stay in fp32.
// Attention is batched over head-groups via blockIdx.z; the P@V GEMM uses a
// 256x64 tile so all 8 waves work when N = head_dim = 64.
// Global->LDS staging uses the gfx1250 async-copy pipe (ASYNCcnt) when the
// toolchain exposes it, else falls back to load+ds_store.
// ---------------------------------------------------------------------------

#define TPB 256

typedef __attribute__((ext_vector_type(16))) _Float16 v16h;
typedef __attribute__((ext_vector_type(8)))  float    v8f;
typedef __attribute__((ext_vector_type(4)))  int      v4i;

union Frag32 { uint4 u[2]; v16h v; };   // 32 bytes = one 16-bit A/B fragment

static constexpr int kB  = 4;
static constexpr int kS  = 2048;
static constexpr int kD  = 1024;
static constexpr int kH  = 16;
static constexpr int kHD = 64;
static constexpr int kF  = 4096;
static constexpr int kL  = 6;
static constexpr int kNT = kB * kS;     // 8192 tokens

#if __has_builtin(__builtin_amdgcn_global_load_async_to_lds_b128) && \
    __has_builtin(__builtin_amdgcn_s_wait_asynccnt)
#define USE_ASYNC_COPY 1
typedef __attribute__((address_space(1))) v4i* gptr_v4i;
typedef __attribute__((address_space(3))) v4i* lptr_v4i;
#else
#define USE_ASYNC_COPY 0
#endif

// 16-byte global -> LDS copy (async pipe when available)
__device__ __forceinline__ void copy_b128_to_lds(const _Float16* gp, _Float16* lp)
{
#if USE_ASYNC_COPY
    __builtin_amdgcn_global_load_async_to_lds_b128(
        (gptr_v4i)(void*)gp, (lptr_v4i)(void*)lp, 0, 0);
#else
    *(uint4*)lp = *(const uint4*)gp;
#endif
}

__device__ __forceinline__ void staging_fence()
{
#if USE_ASYNC_COPY
    __builtin_amdgcn_s_wait_asynccnt(0);
#endif
    __syncthreads();
}

// ------------------------------- GEMM --------------------------------------
// C[M,N] = A[M,K] * B + bias, A f16 row-major (lda), B f16:
//   BT=false : B row-major K x N (ldb)
//   BT=true  : B given as Bt row-major N x K (ldb)  =>  C = A * Bt^T
// Per-z batch: A += z*sA, B += z*sB, C += z*sC (elements).
// Block 256 threads = 8 waves; tile BMT x BNT x 32;
//   wave tile 64x32 -> waves along N = BNT/32, along M = 8/(BNT/32).
#define BKK 32
#define APITCH 40   // LDS row pitch in halves (80B: 16B-aligned, bank-skewed)
#define BPITCH 40

template <bool BT, int BMT, int BNT>
__global__ __launch_bounds__(TPB) void wmma_gemm(
    int M, int N, int K,
    const _Float16* __restrict__ A, int lda, size_t sA,
    const _Float16* __restrict__ Bm, int ldb, size_t sB,
    const float* __restrict__ bias,
    float* __restrict__ outF, _Float16* __restrict__ outH, int ldc, size_t sC,
    int relu)
{
    constexpr int WN   = BNT / 32;       // waves along N (2 or 4)
    constexpr int AIT  = BMT / 64;       // A-tile load iterations
    constexpr int BIT  = BNT / 64;       // B-tile load iterations
    constexpr int NCH  = BNT / 8;        // 8-half chunks per B row (NN path)

    __shared__ _Float16 As[BMT * APITCH];
    __shared__ _Float16 Bs[BNT * BPITCH];

    const int z = blockIdx.z;
    A  += (size_t)z * sA;
    Bm += (size_t)z * sB;
    if (outF) outF += (size_t)z * sC;
    if (outH) outH += (size_t)z * sC;

    const int tid  = threadIdx.x;
    const int lane = tid & 31;
    const int wave = tid >> 5;
    const int m0 = blockIdx.y * BMT;
    const int n0 = blockIdx.x * BNT;
    const int wm = (wave / WN) * 64;
    const int wn = (wave % WN) * 32;

    v8f acc[4][2];
#pragma unroll
    for (int i = 0; i < 4; ++i)
#pragma unroll
        for (int j = 0; j < 2; ++j) acc[i][j] = (v8f)0.0f;

    const int mrow  = lane & 15;
    const int khalf = lane >> 4;

    for (int k0 = 0; k0 < K; k0 += BKK) {
        // ---- stage A tile: BMT x 32 halves, (BMT*4) 8-half chunks ----
#pragma unroll
        for (int it = 0; it < AIT; ++it) {
            int c   = tid + it * TPB;
            int row = c >> 2;                  // 0..BMT-1
            int kc  = (c & 3) * 8;             // 0,8,16,24
            int gm  = m0 + row;
            if (gm < M) {
                const _Float16* gp = A + (size_t)gm * lda + k0 + kc;
                copy_b128_to_lds(gp, &As[row * APITCH + kc]);
                if (k0 + BKK < K) __builtin_prefetch(gp + BKK, 0, 1); // global_prefetch_b8
            } else {
                uint4 zv = {0u, 0u, 0u, 0u};
                *(uint4*)&As[row * APITCH + kc] = zv;
            }
        }
        // ---- stage B tile into Bs[n][k] layout ----
        if (BT) {
            // Bt is N x K row-major: contiguous in k -> b128 copies
#pragma unroll
            for (int it = 0; it < BIT; ++it) {
                int c   = tid + it * TPB;
                int row = c >> 2;              // n 0..BNT-1
                int kc  = (c & 3) * 8;
                int gn  = n0 + row;
                if (gn < N) {
                    copy_b128_to_lds(Bm + (size_t)gn * ldb + k0 + kc,
                                     &Bs[row * BPITCH + kc]);
                } else {
                    uint4 zv = {0u, 0u, 0u, 0u};
                    *(uint4*)&Bs[row * BPITCH + kc] = zv;
                }
            }
        } else {
            // B is K x N row-major: transpose while storing to LDS
#pragma unroll
            for (int it = 0; it < BIT; ++it) {
                int c  = tid + it * TPB;
                int k  = c / NCH;              // 0..31
                int nc = (c % NCH) * 8;        // 0..BNT-8
                int gn = n0 + nc;
                uint4 val = {0u, 0u, 0u, 0u};
                if (gn < N) val = *(const uint4*)(Bm + (size_t)(k0 + k) * ldb + gn);
                const _Float16* hv = (const _Float16*)&val;
#pragma unroll
                for (int j = 0; j < 8; ++j) Bs[(nc + j) * BPITCH + k] = hv[j];
            }
        }
        staging_fence();   // s_wait_asynccnt 0 (if async) + barrier

        // ---- load fragments (ISA 16-bit A 16x32 / B 32x16 layouts) ----
        Frag32 afr[4], bfr[2];
#pragma unroll
        for (int mi = 0; mi < 4; ++mi) {
            const _Float16* base = &As[(wm + mi * 16 + mrow) * APITCH + 8 * khalf];
            afr[mi].u[0] = *(const uint4*)(base);        // K = 8*khalf + 0..7
            afr[mi].u[1] = *(const uint4*)(base + 16);   // K = 16 + 8*khalf + 0..7
        }
#pragma unroll
        for (int ni = 0; ni < 2; ++ni) {
            const _Float16* base = &Bs[(wn + ni * 16 + mrow) * BPITCH + 8 * khalf];
            bfr[ni].u[0] = *(const uint4*)(base);        // K = 8*khalf + 0..7
            bfr[ni].u[1] = *(const uint4*)(base + 16);   // K = 16 + 8*khalf + 0..7
        }

#pragma unroll
        for (int mi = 0; mi < 4; ++mi)
#pragma unroll
            for (int ni = 0; ni < 2; ++ni)
                acc[mi][ni] = __builtin_amdgcn_wmma_f32_16x16x32_f16(
                    false, afr[mi].v, false, bfr[ni].v,
                    (short)0, acc[mi][ni], false, false);
        __syncthreads();
    }

    // ---- epilogue: C/D layout VGPR r -> M = r + 8*(lane>=16), N = lane&15 ----
#pragma unroll
    for (int mi = 0; mi < 4; ++mi) {
#pragma unroll
        for (int ni = 0; ni < 2; ++ni) {
#pragma unroll
            for (int r = 0; r < 8; ++r) {
                int gm = m0 + wm + mi * 16 + r + 8 * khalf;
                int gn = n0 + wn + ni * 16 + mrow;
                if (gm < M && gn < N) {
                    float v = acc[mi][ni][r];
                    if (bias) v += bias[gn];
                    if (relu) v = v > 0.0f ? v : 0.0f;
                    size_t idx = (size_t)gm * ldc + gn;
                    if (outF) outF[idx] = v;
                    if (outH) outH[idx] = (_Float16)v;
                }
            }
        }
    }
}

// ------------------------------ helpers ------------------------------------

__global__ __launch_bounds__(TPB) void cvt_f32_f16(
    const float* __restrict__ in, _Float16* __restrict__ out, size_t n)
{
    size_t i = (size_t)blockIdx.x * TPB + threadIdx.x;
    size_t stride = (size_t)gridDim.x * TPB;
    for (; i < n; i += stride) out[i] = (_Float16)in[i];
}

__global__ __launch_bounds__(TPB) void embed_kernel(
    const int* __restrict__ src, const float* __restrict__ tok,
    const float* __restrict__ pos,
    float* __restrict__ x, _Float16* __restrict__ xh)
{
    int t = blockIdx.x;                 // token 0..NT-1
    int s = t & (kS - 1);               // position in sequence
    int id = src[t];
    size_t base = (size_t)t * kD;
    for (int d = threadIdx.x; d < kD; d += TPB) {
        float v = tok[(size_t)id * kD + d] * 32.0f + pos[(size_t)s * kD + d];
        x[base + d]  = v;
        xh[base + d] = (_Float16)v;
    }
}

// grid (S, Z): row softmax of scores[z] -> f16 probs[z]
__global__ __launch_bounds__(TPB) void softmax_kernel(
    const float* __restrict__ scores, const int* __restrict__ mask,
    _Float16* __restrict__ P, int S, float scale)
{
    __shared__ float red[TPB];
    int row = blockIdx.x;
    int z   = blockIdx.y;
    const float* srow = scores + ((size_t)z * S + row) * S;
    _Float16*    prow = P      + ((size_t)z * S + row) * S;
    int tid = threadIdx.x;

    float mx = -1e30f;
    for (int k = tid; k < S; k += TPB) {
        float v = mask[k] ? srow[k] * scale : -1e10f;
        mx = fmaxf(mx, v);
    }
    red[tid] = mx; __syncthreads();
    for (int o = TPB / 2; o > 0; o >>= 1) {
        if (tid < o) red[tid] = fmaxf(red[tid], red[tid + o]);
        __syncthreads();
    }
    mx = red[0]; __syncthreads();

    float sum = 0.0f;
    for (int k = tid; k < S; k += TPB) {
        float v = mask[k] ? srow[k] * scale : -1e10f;
        sum += __expf(v - mx);
    }
    red[tid] = sum; __syncthreads();
    for (int o = TPB / 2; o > 0; o >>= 1) {
        if (tid < o) red[tid] += red[tid + o];
        __syncthreads();
    }
    float inv = 1.0f / red[0];

    for (int k = tid; k < S; k += TPB) {
        float v = mask[k] ? srow[k] * scale : -1e10f;
        prow[k] = (_Float16)(__expf(v - mx) * inv);
    }
}

// x = LN(x + y) * g + b ; also writes f16 copy of x
__global__ __launch_bounds__(TPB) void add_ln_kernel(
    float* __restrict__ x, _Float16* __restrict__ xh,
    const float* __restrict__ y,
    const float* __restrict__ g, const float* __restrict__ bta)
{
    __shared__ float red[TPB];
    int tid = threadIdx.x;
    size_t base = (size_t)blockIdx.x * kD;

    float v[4];
    float s = 0.0f;
#pragma unroll
    for (int j = 0; j < 4; ++j) {
        int d = tid + j * TPB;
        v[j] = x[base + d] + y[base + d];
        s += v[j];
    }
    red[tid] = s; __syncthreads();
    for (int o = TPB / 2; o > 0; o >>= 1) {
        if (tid < o) red[tid] += red[tid + o];
        __syncthreads();
    }
    float mu = red[0] * (1.0f / kD);
    __syncthreads();

    s = 0.0f;
#pragma unroll
    for (int j = 0; j < 4; ++j) { float dv = v[j] - mu; s += dv * dv; }
    red[tid] = s; __syncthreads();
    for (int o = TPB / 2; o > 0; o >>= 1) {
        if (tid < o) red[tid] += red[tid + o];
        __syncthreads();
    }
    float rstd = rsqrtf(red[0] * (1.0f / kD) + 1e-5f);

#pragma unroll
    for (int j = 0; j < 4; ++j) {
        int d = tid + j * TPB;
        float o = (v[j] - mu) * rstd * g[d] + bta[d];
        x[base + d]  = o;
        xh[base + d] = (_Float16)o;
    }
}

// ------------------------------ host ---------------------------------------

static inline void launch_gemm_nn(int M, int N, int K,
    const _Float16* A, int lda, const _Float16* Bm, int ldb,
    const float* bias, float* oF, _Float16* oH, int ldc, int relu,
    hipStream_t stream)
{
    dim3 grid((N + 127) / 128, (M + 127) / 128, 1);
    wmma_gemm<false, 128, 128><<<grid, TPB, 0, stream>>>(
        M, N, K, A, lda, 0, Bm, ldb, 0, bias, oF, oH, ldc, 0, relu);
}

extern "C" void kernel_launch(void* const* d_in, const int* in_sizes, int n_in,
                              void* d_out, int out_size, void* d_ws, size_t ws_size,
                              hipStream_t stream)
{
    (void)in_sizes; (void)n_in; (void)out_size;

    const int*   src      = (const int*)  d_in[0];
    const int*   src_mask = (const int*)  d_in[1];
    const float* tok_emb  = (const float*)d_in[2];
    const float* pos_emb  = (const float*)d_in[3];
    const float* Wq = (const float*)d_in[4];
    const float* bq = (const float*)d_in[5];
    const float* Wk = (const float*)d_in[6];
    const float* bk = (const float*)d_in[7];
    const float* Wv = (const float*)d_in[8];
    const float* bv = (const float*)d_in[9];
    const float* Wo = (const float*)d_in[10];
    const float* bo = (const float*)d_in[11];
    const float* ln1_g = (const float*)d_in[12];
    const float* ln1_b = (const float*)d_in[13];
    const float* W1 = (const float*)d_in[14];
    const float* b1 = (const float*)d_in[15];
    const float* W2 = (const float*)d_in[16];
    const float* b2 = (const float*)d_in[17];
    const float* ln2_g = (const float*)d_in[18];
    const float* ln2_b = (const float*)d_in[19];

    // ---- workspace layout (~352 MB base + attention scratch) ----
    char* p = (char*)d_ws;
    auto alloc = [&](size_t bytes) -> char* {
        char* r = p; p += (bytes + 255) & ~(size_t)255; return r;
    };
    float*    xf     = (float*)   alloc((size_t)kNT * kD * 4);
    _Float16* xh     = (_Float16*)alloc((size_t)kNT * kD * 2);
    _Float16* qh     = (_Float16*)alloc((size_t)kNT * kD * 2);
    _Float16* kh     = (_Float16*)alloc((size_t)kNT * kD * 2);
    _Float16* vh     = (_Float16*)alloc((size_t)kNT * kD * 2);
    _Float16* attnh  = (_Float16*)alloc((size_t)kNT * kD * 2);
    _Float16* ffh    = (_Float16*)alloc((size_t)kNT * kF * 2);
    float*    yf     = (float*)   alloc((size_t)kNT * kD * 4);
    _Float16* wq16   = (_Float16*)alloc((size_t)kL * kD * kD * 2);
    _Float16* wk16   = (_Float16*)alloc((size_t)kL * kD * kD * 2);
    _Float16* wv16   = (_Float16*)alloc((size_t)kL * kD * kD * 2);
    _Float16* wo16   = (_Float16*)alloc((size_t)kL * kD * kD * 2);
    _Float16* w116   = (_Float16*)alloc((size_t)kL * kD * kF * 2);
    _Float16* w216   = (_Float16*)alloc((size_t)kL * kF * kD * 2);

    // Head-group size G: batch G heads per attention launch if scratch fits.
    size_t used = (size_t)(p - (char*)d_ws);
    int G = 1;
    for (int g = kH; g >= 1; g >>= 1) {
        size_t need = (size_t)g * kS * kS * 6 + 1024;   // f32 scores + f16 probs
        if (used + need <= ws_size) { G = g; break; }
    }
    float*    scores = (float*)   alloc((size_t)G * kS * kS * 4);
    _Float16* ph     = (_Float16*)alloc((size_t)G * kS * kS * 2);

    // ---- convert weights to f16 ----
    size_t nw = (size_t)kL * kD * kD;
    size_t nf = (size_t)kL * kD * kF;
    cvt_f32_f16<<<2048, TPB, 0, stream>>>(Wq, wq16, nw);
    cvt_f32_f16<<<2048, TPB, 0, stream>>>(Wk, wk16, nw);
    cvt_f32_f16<<<2048, TPB, 0, stream>>>(Wv, wv16, nw);
    cvt_f32_f16<<<2048, TPB, 0, stream>>>(Wo, wo16, nw);
    cvt_f32_f16<<<2048, TPB, 0, stream>>>(W1, w116, nf);
    cvt_f32_f16<<<2048, TPB, 0, stream>>>(W2, w216, nf);

    // ---- embedding ----
    embed_kernel<<<kNT, TPB, 0, stream>>>(src, tok_emb, pos_emb, xf, xh);

    const float scale = 0.125f;   // 1/sqrt(HD=64)

    for (int i = 0; i < kL; ++i) {
        const _Float16* wq = wq16 + (size_t)i * kD * kD;
        const _Float16* wk = wk16 + (size_t)i * kD * kD;
        const _Float16* wv = wv16 + (size_t)i * kD * kD;
        const _Float16* wo = wo16 + (size_t)i * kD * kD;
        const _Float16* w1 = w116 + (size_t)i * kD * kF;
        const _Float16* w2 = w216 + (size_t)i * kF * kD;

        // Q,K,V = x @ W + b   (f16 outputs)
        launch_gemm_nn(kNT, kD, kD, xh, kD, wq, kD, bq + (size_t)i * kD,
                       nullptr, qh, kD, 0, stream);
        launch_gemm_nn(kNT, kD, kD, xh, kD, wk, kD, bk + (size_t)i * kD,
                       nullptr, kh, kD, 0, stream);
        launch_gemm_nn(kNT, kD, kD, xh, kD, wv, kD, bv + (size_t)i * kD,
                       nullptr, vh, kD, 0, stream);

        // attention: G heads per launch via blockIdx.z (head h = hg + z)
        for (int b = 0; b < kB; ++b) {
            for (int hg = 0; hg < kH; hg += G) {
                const size_t off = (size_t)b * kS * kD + (size_t)hg * kHD;
                const _Float16* qs = qh + off;
                const _Float16* ks = kh + off;
                const _Float16* vs = vh + off;
                _Float16*       as = attnh + off;

                // scores[z] = Q_z @ K_z^T   (S x S, f32)
                dim3 g1(kS / 128, kS / 128, G);
                wmma_gemm<true, 128, 128><<<g1, TPB, 0, stream>>>(
                    kS, kS, kHD,
                    qs, kD, (size_t)kHD,
                    ks, kD, (size_t)kHD,
                    nullptr, scores, nullptr, kS, (size_t)kS * kS, 0);

                // P[z] = softmax(scores[z] * scale, mask)  -> f16
                softmax_kernel<<<dim3(kS, G), TPB, 0, stream>>>(
                    scores, src_mask + (size_t)b * kS, ph, kS, scale);

                // attn[z] = P[z] @ V_z   (S x 64, f16 into head slice)
                dim3 g3(1, kS / 256, G);
                wmma_gemm<false, 256, 64><<<g3, TPB, 0, stream>>>(
                    kS, kHD, kS,
                    ph, kS, (size_t)kS * kS,
                    vs, kD, (size_t)kHD,
                    nullptr, nullptr, as, kD, (size_t)kHD, 0);
            }
        }

        // y = attn @ Wo + bo ; x = LN(x + y)
        launch_gemm_nn(kNT, kD, kD, attnh, kD, wo, kD, bo + (size_t)i * kD,
                       yf, nullptr, kD, 0, stream);
        add_ln_kernel<<<kNT, TPB, 0, stream>>>(
            xf, xh, yf, ln1_g + (size_t)i * kD, ln1_b + (size_t)i * kD);

        // h = relu(x @ W1 + b1) ; y = h @ W2 + b2 ; x = LN(x + y)
        launch_gemm_nn(kNT, kF, kD, xh, kD, w1, kF, b1 + (size_t)i * kF,
                       nullptr, ffh, kF, 1, stream);
        launch_gemm_nn(kNT, kD, kF, ffh, kF, w2, kD, b2 + (size_t)i * kD,
                       yf, nullptr, kD, 0, stream);
        add_ln_kernel<<<kNT, TPB, 0, stream>>>(
            xf, xh, yf, ln2_g + (size_t)i * kD, ln2_b + (size_t)i * kD);
    }

    // ---- output (fp32) ----
    (void)hipMemcpyAsync(d_out, xf, (size_t)kNT * kD * 4,
                         hipMemcpyDeviceToDevice, stream);
}
